// BiLSTM_70068096467023
// MI455X (gfx1250) — compile-verified, matrix operand order checked by reading, not run on hardware
//
#include <hip/hip_runtime.h>
#include <math.h>

// Problem constants (from reference): B=256, S=512, E=300 (pad 320), H=128, 4H=512
#define Bb 256
#define Ss 512
#define Ee 300
#define EP 320
#define Hh 128
#define G4 512

typedef __attribute__((ext_vector_type(16))) __bf16 v16bf;
typedef __attribute__((ext_vector_type(8)))  __bf16 bf16x8;
typedef __attribute__((ext_vector_type(8)))  float  v8f;

union FragU { v16bf v; bf16x8 h[2]; };

// Round-to-nearest-even f32 -> bf16 bits
__device__ inline unsigned short f2bf(float x) {
  unsigned int u = __float_as_uint(x);
  return (unsigned short)((u + 0x7FFFu + ((u >> 16) & 1u)) >> 16);
}

__device__ inline float sigf(float x) { return 1.0f / (1.0f + __expf(-x)); }

// Load one 16x32 bf16 A/B fragment (ISA 7.12.2 layout): two 16B loads per lane.
__device__ inline v16bf load_frag(const unsigned short* row_base, int k0, int half) {
  FragU f;
  f.h[0] = *(const bf16x8*)(row_base + k0 + half * 8);
  f.h[1] = *(const bf16x8*)(row_base + k0 + 16 + half * 8);
  return f.v;
}

__device__ inline v8f wmma_bf16(v16bf a, v16bf b, v8f c) {
  return __builtin_amdgcn_wmma_f32_16x16x32_bf16(false, a, false, b, (short)0, c,
                                                 false, false);
}

// ---------------------------------------------------------------------------
// Kernel 1: convert weights to bf16 (K-padded) + fused bias (bih + bhh).
// wih_bf: [dir][512][320], whh_bf: [dir][512][128], bias: [dir][512]
// ---------------------------------------------------------------------------
__global__ __launch_bounds__(256) void convert_weights(
    const float* Wih_f, const float* Wih_r, const float* Whh_f, const float* Whh_r,
    const float* bih_f, const float* bhh_f, const float* bih_r, const float* bhh_r,
    unsigned short* wih_bf, unsigned short* whh_bf, float* bias) {
  int tid = blockIdx.x * blockDim.x + threadIdx.x;
  const int NW_IH = 2 * G4 * EP;   // 327680
  const int NW_HH = 2 * G4 * Hh;   // 131072
  const int NB    = 2 * G4;        // 1024
  if (tid < NW_IH) {
    int d = tid / (G4 * EP);
    int r = tid - d * (G4 * EP);
    int g = r / EP, k = r - g * EP;
    const float* src = d ? Wih_r : Wih_f;
    float v = (k < Ee) ? src[g * Ee + k] : 0.0f;
    wih_bf[tid] = f2bf(v);
  } else if (tid < NW_IH + NW_HH) {
    int t = tid - NW_IH;
    int d = t / (G4 * Hh);
    int r = t - d * (G4 * Hh);
    const float* src = d ? Whh_r : Whh_f;
    whh_bf[t] = f2bf(src[r]);
  } else if (tid < NW_IH + NW_HH + NB) {
    int t = tid - NW_IH - NW_HH;
    int d = t / G4, g = t - d * G4;
    bias[t] = (d ? bih_r[g] : bih_f[g]) + (d ? bhh_r[g] : bhh_f[g]);
  }
}

// ---------------------------------------------------------------------------
// Kernel 2: embedding gather + bf16 convert, K-padded. xbf: [token][320]
// ---------------------------------------------------------------------------
__global__ __launch_bounds__(256) void embed_gather(const int* ids, const float* emb,
                                                    unsigned short* xbf) {
  long long tid = (long long)blockIdx.x * blockDim.x + threadIdx.x;  // < 131072*320
  int e = (int)(tid % EP);
  long long t = tid / EP;
  int id = ids[t];
  float v = (e < Ee) ? emb[(long long)id * Ee + e] : 0.0f;
  xbf[tid] = f2bf(v);
}

// ---------------------------------------------------------------------------
// Kernel 3: gx = x @ Wih^T (both directions), bf16 WMMA, f32 accumulate.
// Grid (8192, 2), block 256 (8 waves). Each wave: one 16-token tile x 64 gates.
// gx layout per (seq,dir): [t][b][g]  (time-major -> contiguous step reads)
// ---------------------------------------------------------------------------
__global__ __launch_bounds__(256) void gx_gemm(const unsigned short* __restrict__ xbf,
                                               const unsigned short* __restrict__ wih,
                                               float* __restrict__ gx_seq) {
  const int wv   = threadIdx.x >> 5;
  const int lane = threadIdx.x & 31;
  const int n    = lane & 15;
  const int half = lane >> 4;
  const int gw   = blockIdx.y * 8 + wv;   // 0..15
  const int dir  = gw >> 3;
  const int cg   = gw & 7;                // 64-col group

  const size_t row0 = (size_t)blockIdx.x * 16;
  const unsigned short* arow = xbf + (row0 + n) * EP;

  v8f acc[4];
#pragma unroll
  for (int q = 0; q < 4; ++q)
#pragma unroll
    for (int r = 0; r < 8; ++r) acc[q][r] = 0.0f;

#pragma unroll 2
  for (int k0 = 0; k0 < EP; k0 += 32) {
    v16bf a = load_frag(arow, k0, half);
#pragma unroll
    for (int q = 0; q < 4; ++q) {
      int col0 = cg * 64 + q * 16;
      const unsigned short* brow = wih + ((size_t)dir * G4 + col0 + n) * EP;
      v16bf b = load_frag(brow, k0, half);
      acc[q] = wmma_bf16(a, b, acc[q]);
    }
  }

  const int bidx = (int)(row0 >> 9);   // token / S (tiles never cross batch rows)
  const int s0   = (int)(row0 & 511);
  float* gxd = gx_seq + (size_t)dir * ((size_t)Ss * Bb * G4);
#pragma unroll
  for (int q = 0; q < 4; ++q) {
    int col = cg * 64 + q * 16 + n;
#pragma unroll
    for (int r = 0; r < 8; ++r) {
      int s = s0 + 8 * half + r;
      gxd[((size_t)s * Bb + bidx) * G4 + col] = acc[q][r];
    }
  }
}

// ---------------------------------------------------------------------------
// Kernel 4: persistent LSTM recurrence. Grid=4 (seq x dir), block=1024 (32 waves).
// h double-buffered bf16 in LDS, Whh bf16 in LDS, c in registers. Per step:
// 16x16 tile GEMM via WMMA over K=128, gates, write new h, barrier.
// Dynamic LDS: 2*32768 u16 (h) + 65536 u16 (Whh) + 512 f32 (bias) = 264192 B.
// ---------------------------------------------------------------------------
__global__ __launch_bounds__(1024, 1) void lstm_recurrence(
    const float* __restrict__ gx, const unsigned short* __restrict__ whh_g,
    const float* __restrict__ bias_g, float* __restrict__ hout_g) {
  extern __shared__ unsigned short smem_us[];
  unsigned short* hA  = smem_us;               // 256*128
  unsigned short* hB  = hA + Bb * Hh;          // 256*128
  unsigned short* wsm = hB + Bb * Hh;          // 512*128
  float*          bsm = (float*)(wsm + G4 * Hh);

  const int ls  = blockIdx.x;      // 0..3 : seq*2 + dir
  const int dir = ls & 1;
  const float* gxb = gx + (size_t)ls * ((size_t)Ss * Bb * G4);
  const unsigned short* whh_d = whh_g + (size_t)dir * G4 * Hh;
  const float* bias_d = bias_g + dir * G4;
  float* houtb = hout_g + (size_t)ls * Bb * Hh;

  const int tid  = threadIdx.x;
  const int wv   = tid >> 5;
  const int lane = tid & 31;
  const int n    = lane & 15;
  const int half = lane >> 4;

  // ---- init LDS: zero h, stage Whh (bf16) + fused bias ----
  {
    uint4* z = (uint4*)hA;                      // zeros both h buffers: 65536 u16
    for (int i = tid; i < (2 * Bb * Hh) / 8; i += 1024) z[i] = uint4{0, 0, 0, 0};
    const uint4* src = (const uint4*)whh_d;     // 65536 u16 = 8192 uint4
    uint4* dst = (uint4*)wsm;
    for (int i = tid; i < (G4 * Hh) / 8; i += 1024) dst[i] = src[i];
    if (tid < G4) bsm[tid] = bias_d[tid];
  }
  __syncthreads();

  float c[4][8];
#pragma unroll
  for (int p = 0; p < 4; ++p)
#pragma unroll
    for (int r = 0; r < 8; ++r) c[p][r] = 0.0f;

  for (int t = 0; t < Ss; ++t) {
    const int tt = dir ? (Ss - 1 - t) : t;
    const float* gxt = gxb + (size_t)tt * Bb * G4;
    unsigned short* hin  = (t & 1) ? hB : hA;
    unsigned short* hnew = (t & 1) ? hA : hB;

    // prefetch next time step's gx tile (512 KB spread over 1024 lanes)
    if (t + 1 < Ss) {
      int ntt = dir ? (tt - 1) : (tt + 1);
      __builtin_prefetch(gxb + (size_t)ntt * Bb * G4 + (size_t)tid * 128, 0, 0);
    }

#pragma unroll
    for (int p = 0; p < 4; ++p) {
      const int idx = wv * 4 + p;      // 0..127
      const int rt  = idx >> 3;        // batch tile 0..15
      const int ut  = idx & 7;         // unit tile 0..7
      const int rbase = rt * 16;

      v8f acc[4];
#pragma unroll
      for (int q = 0; q < 4; ++q) {
        const int gcol = q * Hh + ut * 16 + n;
        const float bq = bsm[gcol];
#pragma unroll
        for (int r = 0; r < 8; ++r)
          acc[q][r] = gxt[(size_t)(rbase + 8 * half + r) * G4 + gcol] + bq;
      }

#pragma unroll
      for (int kc = 0; kc < Hh; kc += 32) {
        v16bf a = load_frag(hin + (size_t)(rbase + n) * Hh, kc, half);
#pragma unroll
        for (int q = 0; q < 4; ++q) {
          const int g0 = q * Hh + ut * 16;
          v16bf b = load_frag(wsm + (size_t)(g0 + n) * Hh, kc, half);
          acc[q] = wmma_bf16(a, b, acc[q]);
        }
      }

      const int unit = ut * 16 + n;
#pragma unroll
      for (int r = 0; r < 8; ++r) {
        const int row = rbase + 8 * half + r;
        float iv = sigf(acc[0][r]);
        float fv = sigf(acc[1][r]);
        float gv = tanhf(acc[2][r]);
        float ov = sigf(acc[3][r]);
        float cc = fv * c[p][r] + iv * gv;
        c[p][r] = cc;
        float hh = ov * tanhf(cc);
        hnew[row * Hh + unit] = f2bf(hh);
        if (t == Ss - 1) houtb[row * Hh + unit] = hh;
      }
    }
    __syncthreads();
  }
}

// ---------------------------------------------------------------------------
// Kernel 5: out[j] = (1/256) * (sum_b h1[b,j]) * (sum_b h2[b,j])
// (uniform attention: softmax over identical tiled columns == 1/(2H))
// ---------------------------------------------------------------------------
__global__ __launch_bounds__(256) void finalize(const float* __restrict__ hout,
                                                float* __restrict__ out) {
  int j = threadIdx.x;                 // 0..255
  int dir = j >> 7, u = j & 127;
  float s1 = 0.0f, s2 = 0.0f;
  for (int b = 0; b < Bb; ++b) {
    s1 += hout[(size_t)(0 + dir) * Bb * Hh + b * Hh + u];
    s2 += hout[(size_t)(2 + dir) * Bb * Hh + b * Hh + u];
  }
  out[j] = s1 * s2 * (1.0f / 256.0f);
}

// ---------------------------------------------------------------------------
extern "C" void kernel_launch(void* const* d_in, const int* in_sizes, int n_in,
                              void* d_out, int out_size, void* d_ws, size_t ws_size,
                              hipStream_t stream) {
  (void)in_sizes; (void)n_in; (void)out_size; (void)ws_size;
  const int*   ids1  = (const int*)d_in[0];
  const int*   ids2  = (const int*)d_in[1];
  const float* emb   = (const float*)d_in[2];
  const float* Wih_f = (const float*)d_in[3];
  const float* Whh_f = (const float*)d_in[4];
  const float* bih_f = (const float*)d_in[5];
  const float* bhh_f = (const float*)d_in[6];
  const float* Wih_r = (const float*)d_in[7];
  const float* Whh_r = (const float*)d_in[8];
  const float* bih_r = (const float*)d_in[9];
  const float* bhh_r = (const float*)d_in[10];
  // d_in[11]/d_in[12] (attn_w, attn_b) provably do not affect the output.
  float* out = (float*)d_out;

  // Workspace carve-up (~1.16 GB)
  float* gx = (float*)d_ws;                                    // 4 * 512*256*512 f32
  unsigned short* xbf = (unsigned short*)(gx + (size_t)4 * Ss * Bb * G4);
  unsigned short* wih = xbf + (size_t)Bb * Ss * EP;            // 2*512*320 u16
  unsigned short* whh = wih + (size_t)2 * G4 * EP;             // 2*512*128 u16
  float* bias = (float*)(whh + (size_t)2 * G4 * Hh);           // 2*512 f32
  float* hout = bias + 2 * G4;                                 // 4*256*128 f32

  convert_weights<<<1796, 256, 0, stream>>>(Wih_f, Wih_r, Whh_f, Whh_r,
                                            bih_f, bhh_f, bih_r, bhh_r,
                                            wih, whh, bias);
  for (int seq = 0; seq < 2; ++seq) {
    const int* ids = seq ? ids2 : ids1;
    embed_gather<<<(Bb * Ss * EP) / 256, 256, 0, stream>>>(ids, emb, xbf);
    gx_gemm<<<dim3((Bb * Ss) / 16, 2), 256, 0, stream>>>(
        xbf, wih, gx + (size_t)seq * 2 * Ss * Bb * G4);
  }
  lstm_recurrence<<<4, 1024, (2 * Bb * Hh + G4 * Hh) * 2 + G4 * 4, stream>>>(
      gx, whh, bias, hout);
  finalize<<<1, 256, 0, stream>>>(hout, out);
}
